// Model_36928128811550
// MI455X (gfx1250) — matrix operand order repaired and checked by reading, not compiled, not run on hardware
//
#include <hip/hip_runtime.h>
#include <hip/hip_bf16.h>
#include <math.h>

// Dims
#define B_   16
#define S_   400
#define E_   256
#define H_   512
#define G_   2048     // 4*H
#define T_   32       // decode steps
#define V_   50000
#define VE_  50100
#define CH3_ 1536     // 3*H

typedef __attribute__((ext_vector_type(2)))  float  v2f;
typedef __attribute__((ext_vector_type(8)))  float  v8f;
typedef __attribute__((ext_vector_type(16))) __bf16 v16bf;

__device__ __forceinline__ float sigm(float x) { return 1.f / (1.f + __expf(-x)); }

// ---------- f32 WMMA (16x16x4) fragments ----------
// A (16xK row-major): lane r holds row r, hi half holds K=k0+2..k0+3.
// B from W[N][K] row-major (X @ W^T): identical indexing with rows = N.
template <typename T>
__device__ __forceinline__ v2f ldfrag(const T* base, int ld, int k0, int r, int hi) {
  const T* p = base + (size_t)r * ld + (k0 + 2 * hi);
  v2f f; f.x = (float)p[0]; f.y = (float)p[1];
  return f;
}

__device__ __forceinline__ v8f wmma4(v2f a, v2f b, v8f c) {
  return __builtin_amdgcn_wmma_f32_16x16x4_f32(false, a, false, b, (short)0, c, false, false);
}

// ---------- bf16 WMMA (16x16x32) fragments ----------
// A (16x32, 16-bit): lane r: K = k0+8*hi .. +7  and  k0+16+8*hi .. +7 (ISA A-layout).
template <typename T>
__device__ __forceinline__ v16bf ldfragA_bf(const T* base, int ld, int k0, int r, int hi) {
  const T* p = base + (size_t)r * ld + k0 + hi * 8;
  v16bf f;
#pragma unroll
  for (int i = 0; i < 8; ++i) f[i] = p[i];
#pragma unroll
  for (int i = 0; i < 8; ++i) f[8 + i] = p[16 + i];
  return f;
}
// B (32x16) from bf16 W[N][K] row-major: lane r = column n0+r, K = k0+8*hi .. +7  and  k0+16+8*hi .. +7.
__device__ __forceinline__ v16bf ldfragB_bf(const __bf16* Wrow, int ld, int k0, int r, int hi) {
  const __bf16* p = Wrow + (size_t)r * ld + k0 + hi * 8;
  v16bf f;
#pragma unroll
  for (int i = 0; i < 8; ++i) f[i] = p[i];
#pragma unroll
  for (int i = 0; i < 8; ++i) f[8 + i] = p[16 + i];
  return f;
}

__device__ __forceinline__ v8f wmma_bf(v16bf a, v16bf b, v8f c) {
  return __builtin_amdgcn_wmma_f32_16x16x32_bf16(false, a, false, b, (short)0, c, false, false);
}

// ---------------- f32 -> bf16 weight conversion ----------------
__global__ void f32_to_bf16(const float* __restrict__ in, __bf16* __restrict__ out, int n) {
  int i = blockIdx.x * 256 + threadIdx.x;
  if (i < n) out[i] = (__bf16)in[i];
}

// ---------------- embedding gathers (emit bf16 activations) ----------------
__global__ void gather_enc(const int* __restrict__ ids, const float* __restrict__ emb,
                           __bf16* __restrict__ out) {
  int idx = blockIdx.x * 256 + threadIdx.x;          // [S][B][E]
  if (idx >= S_ * B_ * E_) return;
  int e = idx & (E_ - 1);
  int bt = idx >> 8;
  int b = bt & (B_ - 1);
  int t = bt >> 4;
  int tok = ids[b * S_ + t];
  if (tok >= V_) tok = 1;                            // UNK
  out[idx] = (__bf16)emb[(size_t)tok * E_ + e];
}

__global__ void gather_dec(const int* __restrict__ tgt, const float* __restrict__ emb,
                           __bf16* __restrict__ out) {
  int idx = blockIdx.x * 256 + threadIdx.x;          // [T][B][E]
  if (idx >= T_ * B_ * E_) return;
  int e = idx & (E_ - 1);
  int bt = idx >> 8;
  int b = bt & (B_ - 1);
  int t = bt >> 4;
  int ti = (t == 0) ? 0 : (t - 1);                   // reference off-by-one
  int tok = tgt[b * 33 + ti];
  if (tok >= V_) tok = 1;
  out[idx] = (__bf16)emb[(size_t)tok * E_ + e];
}

// ---------------- persistent bi-LSTM encoder, bf16 WMMA (grid.x = 2 dirs) -----------
// Each of 32 waves owns one 16-column slice of all four gate blocks; gates + cell
// state live in registers; only h (bf16, 16 KB) round-trips through LDS per step.
__global__ __launch_bounds__(1024) void lstm_encoder(
    const __bf16* __restrict__ x,   // [S][B][E] bf16
    const __bf16* Wf_ih, const __bf16* Wf_hh, const float* bf, float* houtf,
    const __bf16* Wb_ih, const __bf16* Wb_hh, const float* bb, float* houtb) {
  const int rev = blockIdx.x;
  const __bf16* Wih  = rev ? Wb_ih : Wf_ih;
  const __bf16* Whh  = rev ? Wb_hh : Wf_hh;
  const float*  bias = rev ? bb    : bf;
  float*        hout = rev ? houtb : houtf;

  __shared__ __bf16 hbf[B_ * H_];
  const int tid = threadIdx.x, wave = tid >> 5, lane = tid & 31, r = lane & 15, hi = lane >> 4;
  for (int i = tid; i < B_ * H_; i += 1024) hbf[i] = (__bf16)0.f;
  v8f c; for (int j = 0; j < 8; ++j) c[j] = 0.f;
  __syncthreads();

  const int n_i = wave * 16, n_f = H_ + wave * 16, n_g = 2 * H_ + wave * 16, n_o = 3 * H_ + wave * 16;
  const float bi = bias[n_i + r], bfv = bias[n_f + r], bg = bias[n_g + r], bo = bias[n_o + r];
  const __bf16* Wi_i = Wih + (size_t)n_i * E_; const __bf16* Wi_f = Wih + (size_t)n_f * E_;
  const __bf16* Wi_g = Wih + (size_t)n_g * E_; const __bf16* Wi_o = Wih + (size_t)n_o * E_;
  const __bf16* Wh_i = Whh + (size_t)n_i * H_; const __bf16* Wh_f = Whh + (size_t)n_f * H_;
  const __bf16* Wh_g = Whh + (size_t)n_g * H_; const __bf16* Wh_o = Whh + (size_t)n_o * H_;

  for (int step = 0; step < S_; ++step) {
    const int t = rev ? (S_ - 1 - step) : step;
    const __bf16* xt = x + (size_t)t * B_ * E_;
    v8f ai, af, ag, ao;
    for (int j = 0; j < 8; ++j) { ai[j] = bi; af[j] = bfv; ag[j] = bg; ao[j] = bo; }
#pragma unroll
    for (int k0 = 0; k0 < E_; k0 += 32) {           // 8 k-steps over x
      v16bf a = ldfragA_bf(xt, E_, k0, r, hi);
      ai = wmma_bf(a, ldfragB_bf(Wi_i, E_, k0, r, hi), ai);
      af = wmma_bf(a, ldfragB_bf(Wi_f, E_, k0, r, hi), af);
      ag = wmma_bf(a, ldfragB_bf(Wi_g, E_, k0, r, hi), ag);
      ao = wmma_bf(a, ldfragB_bf(Wi_o, E_, k0, r, hi), ao);
    }
#pragma unroll 4
    for (int k0 = 0; k0 < H_; k0 += 32) {           // 16 k-steps over h
      v16bf a = ldfragA_bf(hbf, H_, k0, r, hi);
      ai = wmma_bf(a, ldfragB_bf(Wh_i, H_, k0, r, hi), ai);
      af = wmma_bf(a, ldfragB_bf(Wh_f, H_, k0, r, hi), af);
      ag = wmma_bf(a, ldfragB_bf(Wh_g, H_, k0, r, hi), ag);
      ao = wmma_bf(a, ldfragB_bf(Wh_o, H_, k0, r, hi), ao);
    }
    __syncthreads();   // all waves done reading hbf
#pragma unroll
    for (int j = 0; j < 8; ++j) {
      float iv = sigm(ai[j]), fv = sigm(af[j]), gv = tanhf(ag[j]), ov = sigm(ao[j]);
      float cv = fv * c[j] + iv * gv; c[j] = cv;
      float hv = ov * tanhf(cv);
      int bb2 = j + 8 * hi, col = wave * 16 + r;
      hbf[bb2 * H_ + col] = (__bf16)hv;
      hout[(size_t)t * B_ * H_ + bb2 * H_ + col] = hv;
    }
    __syncthreads();   // h fully updated before next step reads it
  }
}

// ---------------- generic f32 WMMA GEMM (tall M, small L2-resident W) ----------------
__global__ __launch_bounds__(256) void gemm_wt(
    const float* __restrict__ A, int lda,
    const float* __restrict__ W, int ldw,
    const float* __restrict__ bias,
    float* __restrict__ D, int ldd,
    int Ntiles, int K, int accumulate) {
  const int wave = threadIdx.x >> 5, lane = threadIdx.x & 31;
  const int r = lane & 15, hi = lane >> 4;
  const int mt = blockIdx.x;
  const int nt = blockIdx.y * 8 + wave;
  if (nt >= Ntiles) return;
  const float* Arow = A + (size_t)mt * 16 * lda;
  const float* Wrow = W + (size_t)nt * 16 * ldw;
  float bv = bias ? bias[nt * 16 + r] : 0.f;
  v8f acc; for (int j = 0; j < 8; ++j) acc[j] = bv;
#pragma unroll 8
  for (int k0 = 0; k0 < K; k0 += 4) {
    v2f a = ldfrag(Arow, lda, k0, r, hi);
    v2f w = ldfrag(Wrow, ldw, k0, r, hi);
    acc = wmma4(a, w, acc);
  }
  float* Drow = D + (size_t)mt * 16 * ldd + nt * 16;
#pragma unroll
  for (int j = 0; j < 8; ++j) {
    int row = j + 8 * hi;
    float v = acc[j];
    if (accumulate) v += Drow[(size_t)row * ldd + r];
    Drow[(size_t)row * ldd + r] = v;
  }
}

// ---------------- strip GEMM for huge W (V1/V2): M fixed = 512 -----------------------
// One block per 16-wide N strip; its 8 waves cover all 32 M-tiles (4 per wave), so
// each W element streams from HBM exactly once (~307 MB for V2 ~= 13 us at 23.3 TB/s).
__global__ __launch_bounds__(256) void gemm_strip(
    const float* __restrict__ A, int lda,   // [512][K]
    const float* __restrict__ W, int ldw,   // [N][K]
    const float* __restrict__ bias,
    float* __restrict__ D, int ldd,
    int K) {
  const int wave = threadIdx.x >> 5, lane = threadIdx.x & 31;
  const int r = lane & 15, hi = lane >> 4;
  const int nt = blockIdx.x;
  const int m0 = wave * 4;                  // this wave's 4 M-tiles
  const float* Wrow = W + (size_t)nt * 16 * ldw;
  float bv = bias ? bias[nt * 16 + r] : 0.f;

  v8f acc[4];
  const float* Ap[4];
#pragma unroll
  for (int m = 0; m < 4; ++m) {
    Ap[m] = A + (size_t)(m0 + m) * 16 * lda;
#pragma unroll
    for (int j = 0; j < 8; ++j) acc[m][j] = bv;
  }

  for (int kk = 0; kk < K; kk += 64) {
    // stream-prefetch the next W k-slice (lane r covers one of the 16 W rows)
    if (kk + 64 < K)
      __builtin_prefetch(Wrow + (size_t)r * ldw + kk + 64, 0, 1);
#pragma unroll 4
    for (int k0 = kk; k0 < kk + 64; k0 += 4) {
      v2f w = ldfrag(Wrow, ldw, k0, r, hi);
      acc[0] = wmma4(ldfrag(Ap[0], lda, k0, r, hi), w, acc[0]);
      acc[1] = wmma4(ldfrag(Ap[1], lda, k0, r, hi), w, acc[1]);
      acc[2] = wmma4(ldfrag(Ap[2], lda, k0, r, hi), w, acc[2]);
      acc[3] = wmma4(ldfrag(Ap[3], lda, k0, r, hi), w, acc[3]);
    }
  }

#pragma unroll
  for (int m = 0; m < 4; ++m) {
    float* Drow = D + (size_t)(m0 + m) * 16 * ldd + nt * 16;
#pragma unroll
    for (int j = 0; j < 8; ++j)
      Drow[(size_t)(j + 8 * hi) * ldd + r] = acc[m][j];
  }
}

// ---------------- enc_proj . v  (per (s,b) row dot) ----------------
__global__ __launch_bounds__(256) void epv_kernel(const float* __restrict__ ep,
                                                  const float* __restrict__ v_w,
                                                  const float* __restrict__ v_b,
                                                  float* __restrict__ epv) {
  int row = blockIdx.x * 8 + (threadIdx.x >> 5);
  int lane = threadIdx.x & 31;
  if (row >= S_ * B_) return;
  const float* p = ep + (size_t)row * H_;
  float s = 0.f;
  for (int j = lane; j < H_; j += 32) s += p[j] * v_w[j];
  for (int off = 16; off; off >>= 1) s += __shfl_xor(s, off, 32);
  if (lane == 0) epv[row] = s + v_b[0];
}

// ---------------- step-invariant attention softmax + closed-form cov_loss -------------
__global__ __launch_bounds__(512) void attn_kernel(const float* __restrict__ epv,
                                                   float* __restrict__ attn,
                                                   float* __restrict__ cov_out) {
  __shared__ float bsum[B_];
  int wave = threadIdx.x >> 5, lane = threadIdx.x & 31;   // 16 waves = 16 batches
  float mx = -1e30f;
  for (int s = lane; s < S_; s += 32) mx = fmaxf(mx, epv[s * B_ + wave]);
  for (int off = 16; off; off >>= 1) mx = fmaxf(mx, __shfl_xor(mx, off, 32));
  float sum = 0.f;
  for (int s = lane; s < S_; s += 32) sum += __expf(epv[s * B_ + wave] - mx);
  for (int off = 16; off; off >>= 1) sum += __shfl_xor(sum, off, 32);
  float inv = 1.f / sum;
  float tot = 0.f;
  for (int s = lane; s < S_; s += 32) {
    float a = __expf(epv[s * B_ + wave] - mx) * inv;
    attn[wave * S_ + s] = a;
    tot += a;
  }
  for (int off = 16; off; off >>= 1) tot += __shfl_xor(tot, off, 32);
  if (lane == 0) bsum[wave] = tot;
  __syncthreads();
  if (threadIdx.x == 0) {
    float cl = 0.f;
    for (int i = 0; i < B_; ++i) cl += bsum[i];
    *cov_out = 31.f * cl;    // sum_{i=1..31} sum min(attn, i*attn) = 31 * sum(attn)
  }
}

// ---------------- constant context = attn @ enc_out ----------------
__global__ void context_kernel(const float* __restrict__ attn,
                               const float* __restrict__ hf,
                               const float* __restrict__ hb,
                               float* __restrict__ context) {
  int idx = blockIdx.x * 256 + threadIdx.x;    // [B][2H]
  if (idx >= B_ * 2 * H_) return;
  int b = idx >> 10, j = idx & 1023;
  const float* src = (j < H_) ? (hf + (size_t)b * H_ + j) : (hb + (size_t)b * H_ + (j - H_));
  float s = 0.f;
  for (int t = 0; t < S_; ++t) s += attn[b * S_ + t] * src[(size_t)t * B_ * H_];
  context[idx] = s;
}

__global__ __launch_bounds__(512) void pgctx_kernel(const float* __restrict__ context,
                                                    const float* __restrict__ wh_w,
                                                    const float* __restrict__ wh_b,
                                                    float* __restrict__ pgctx) {
  int wave = threadIdx.x >> 5, lane = threadIdx.x & 31;
  float s = 0.f;
  const float* cb = context + (size_t)wave * 2 * H_;
  for (int j = lane; j < 2 * H_; j += 32) s += cb[j] * wh_w[j];
  for (int off = 16; off; off >>= 1) s += __shfl_xor(s, off, 32);
  if (lane == 0) pgctx[wave] = s + wh_b[0];
}

// ---------------- duplicate-token pointer distribution (step-invariant) --------------
__global__ void counts_kernel(const int* __restrict__ input, int* __restrict__ counts) {
  int i = blockIdx.x * 256 + threadIdx.x;
  if (i < B_ * S_) atomicAdd(&counts[input[i]], 1);
}

__global__ void rowtok_kernel(const int* __restrict__ input, const float* __restrict__ attn,
                              float* __restrict__ row_tok) {
  int i = blockIdx.x * 256 + threadIdx.x;
  if (i >= B_ * S_) return;
  int b = i / S_, s = i % S_;
  atomicAdd(&row_tok[(size_t)b * VE_ + input[i]], attn[b * S_ + s]);
}

__global__ void pcopy_kernel(const int* __restrict__ input, const int* __restrict__ counts,
                             const float* __restrict__ attn, const float* __restrict__ row_tok,
                             float* __restrict__ p_copy) {
  int i = blockIdx.x * 256 + threadIdx.x;
  if (i >= B_ * S_) return;
  int b = i / S_, s = i % S_;
  int tok = input[i];
  float a = attn[b * S_ + s];
  bool dup = (counts[tok] > 1) && (tok != 0);
  float amod = dup ? a * row_tok[(size_t)b * VE_ + tok] : a;
  atomicAdd(&p_copy[(size_t)b * VE_ + tok], amod);
}

// ---------------- persistent decoder LSTM (bf16 WMMA) + p_gen + hidden_all ----------
__global__ __launch_bounds__(1024) void decoder_seq(
    const __bf16* __restrict__ dec_x, const __bf16* __restrict__ Wih,
    const __bf16* __restrict__ Whh, const float* __restrict__ bias,
    const float* __restrict__ ws_w, const float* __restrict__ ws_b,
    const float* __restrict__ wx_w, const float* __restrict__ wx_b,
    const float* __restrict__ pgctx, const float* __restrict__ context,
    float* __restrict__ hidden_all, float* __restrict__ pgen_all) {
  __shared__ __bf16 hbf[B_ * H_];
  __shared__ float  hf32[B_ * H_];
  const int tid = threadIdx.x, wave = tid >> 5, lane = tid & 31, r = lane & 15, hi = lane >> 4;
  for (int i = tid; i < B_ * H_; i += 1024) { hbf[i] = (__bf16)0.f; hf32[i] = 0.f; }
  v8f c; for (int j = 0; j < 8; ++j) c[j] = 0.f;
  __syncthreads();

  const int n_i = wave * 16, n_f = H_ + wave * 16, n_g = 2 * H_ + wave * 16, n_o = 3 * H_ + wave * 16;
  const float bi = bias[n_i + r], bfv = bias[n_f + r], bg = bias[n_g + r], bo = bias[n_o + r];
  const __bf16* Wi_i = Wih + (size_t)n_i * E_; const __bf16* Wi_f = Wih + (size_t)n_f * E_;
  const __bf16* Wi_g = Wih + (size_t)n_g * E_; const __bf16* Wi_o = Wih + (size_t)n_o * E_;
  const __bf16* Wh_i = Whh + (size_t)n_i * H_; const __bf16* Wh_f = Whh + (size_t)n_f * H_;
  const __bf16* Wh_g = Whh + (size_t)n_g * H_; const __bf16* Wh_o = Whh + (size_t)n_o * H_;

  for (int t = 0; t < T_; ++t) {
    const __bf16* xt = dec_x + (size_t)t * B_ * E_;
    v8f ai, af, ag, ao;
    for (int j = 0; j < 8; ++j) { ai[j] = bi; af[j] = bfv; ag[j] = bg; ao[j] = bo; }
#pragma unroll
    for (int k0 = 0; k0 < E_; k0 += 32) {
      v16bf a = ldfragA_bf(xt, E_, k0, r, hi);
      ai = wmma_bf(a, ldfragB_bf(Wi_i, E_, k0, r, hi), ai);
      af = wmma_bf(a, ldfragB_bf(Wi_f, E_, k0, r, hi), af);
      ag = wmma_bf(a, ldfragB_bf(Wi_g, E_, k0, r, hi), ag);
      ao = wmma_bf(a, ldfragB_bf(Wi_o, E_, k0, r, hi), ao);
    }
#pragma unroll 4
    for (int k0 = 0; k0 < H_; k0 += 32) {
      v16bf a = ldfragA_bf(hbf, H_, k0, r, hi);
      ai = wmma_bf(a, ldfragB_bf(Wh_i, H_, k0, r, hi), ai);
      af = wmma_bf(a, ldfragB_bf(Wh_f, H_, k0, r, hi), af);
      ag = wmma_bf(a, ldfragB_bf(Wh_g, H_, k0, r, hi), ag);
      ao = wmma_bf(a, ldfragB_bf(Wh_o, H_, k0, r, hi), ao);
    }
    __syncthreads();
#pragma unroll
    for (int j = 0; j < 8; ++j) {
      float iv = sigm(ai[j]), fv = sigm(af[j]), gv = tanhf(ag[j]), ov = sigm(ao[j]);
      float cv = fv * c[j] + iv * gv; c[j] = cv;
      float hv = ov * tanhf(cv);
      int bb2 = j + 8 * hi, col = wave * 16 + r;
      hbf[bb2 * H_ + col]  = (__bf16)hv;
      hf32[bb2 * H_ + col] = hv;
    }
    __syncthreads();
    // p_gen (context & wh part precomputed in pgctx)
    if (wave < B_) {
      int bb2 = wave;
      float p = 0.f;
      for (int j = lane; j < H_; j += 32) p += hf32[bb2 * H_ + j] * ws_w[j];
      const __bf16* xb = xt + (size_t)bb2 * E_;
      for (int j = lane; j < E_; j += 32) p += (float)xb[j] * wx_w[j];
      for (int off = 16; off; off >>= 1) p += __shfl_xor(p, off, 32);
      if (lane == 0)
        pgen_all[t * B_ + bb2] = sigm(p + pgctx[bb2] + ws_b[0] + wx_b[0]);
    }
    // hidden_all[(t*B+b)][0:512]=state, [512:1536]=context (const)
    for (int idx = tid; idx < B_ * CH3_; idx += 1024) {
      int bb2 = idx / CH3_, j = idx % CH3_;
      float val = (j < H_) ? hf32[bb2 * H_ + j] : context[bb2 * 2 * H_ + (j - H_)];
      hidden_all[((size_t)t * B_ + bb2) * CH3_ + j] = val;
    }
    // next iteration's h-write is protected by its own pre-update barrier
  }
}

// ---------------- row softmax stats over V2 logits ----------------
__global__ __launch_bounds__(256) void rowstat_kernel(const float* __restrict__ logits,
                                                      float* __restrict__ rowstat) {
  __shared__ float red[256];
  int row = blockIdx.x;
  const float* p = logits + (size_t)row * V_;
  float mx = -1e30f;
  for (int j = threadIdx.x; j < V_; j += 256) mx = fmaxf(mx, p[j]);
  red[threadIdx.x] = mx; __syncthreads();
  for (int s = 128; s; s >>= 1) { if (threadIdx.x < s) red[threadIdx.x] = fmaxf(red[threadIdx.x], red[threadIdx.x + s]); __syncthreads(); }
  mx = red[0]; __syncthreads();
  float sum = 0.f;
  for (int j = threadIdx.x; j < V_; j += 256) sum += __expf(p[j] - mx);
  red[threadIdx.x] = sum; __syncthreads();
  for (int s = 128; s; s >>= 1) { if (threadIdx.x < s) red[threadIdx.x] += red[threadIdx.x + s]; __syncthreads(); }
  if (threadIdx.x == 0) { rowstat[row * 2] = mx; rowstat[row * 2 + 1] = 1.f / red[0]; }
}

// ---------------- final combine: out[b][t][v] ----------------
__global__ void combine_kernel(const float* __restrict__ logits, const float* __restrict__ rowstat,
                               const float* __restrict__ p_copy, const float* __restrict__ pgen_all,
                               float* __restrict__ out) {
  size_t idx = (size_t)blockIdx.x * 256 + threadIdx.x;
  size_t total = (size_t)B_ * T_ * VE_;
  if (idx >= total) return;
  int v = (int)(idx % VE_);
  size_t bt = idx / VE_;
  int t = (int)(bt % T_);
  int b = (int)(bt / T_);
  int row = t * B_ + b;
  float pg = pgen_all[row];
  float pv;
  if (v < V_) pv = __expf(logits[(size_t)row * V_ + v] - rowstat[row * 2]) * rowstat[row * 2 + 1];
  else        pv = 1.0f / (float)V_;
  out[idx] = pv * pg + p_copy[(size_t)b * VE_ + v] * (1.f - pg);
}

// =====================================================================================
extern "C" void kernel_launch(void* const* d_in, const int* in_sizes, int n_in,
                              void* d_out, int out_size, void* d_ws, size_t ws_size,
                              hipStream_t stream) {
  (void)in_sizes; (void)n_in; (void)ws_size;
  // JAX pytree (sorted dict keys) flattening of setup_inputs():
  const int*   input  = (const int*)  d_in[0];   // [16,400]
  const int*   target = (const int*)  d_in[1];   // [16,33]
  const float* V1_w   = (const float*)d_in[2];   // [1536,1536]
  const float* V1_b   = (const float*)d_in[3];
  const float* V2_w   = (const float*)d_in[4];   // [50000,1536]
  const float* V2_b   = (const float*)d_in[5];
  /* Wc_w = d_in[6], Wc_b = d_in[7]: cancel in softmax (no tanh in reference) */
  const float* Wh_w   = (const float*)d_in[8];   // [512,1024]
  const float* Wh_b   = (const float*)d_in[9];
  /* Ws_w = d_in[10], Ws_b = d_in[11]: cancel in softmax */
  const float* dWhh   = (const float*)d_in[12];  // dec.Whh [2048,512]
  const float* dWih   = (const float*)d_in[13];  // dec.Wih [2048,256]
  const float* db     = (const float*)d_in[14];
  const float* emb    = (const float*)d_in[15];  // [50000,256]
  const float* ebWhh  = (const float*)d_in[16];  // enc_b
  const float* ebWih  = (const float*)d_in[17];
  const float* ebb    = (const float*)d_in[18];
  const float* efWhh  = (const float*)d_in[19];  // enc_f
  const float* efWih  = (const float*)d_in[20];
  const float* efb    = (const float*)d_in[21];
  const float* v_w    = (const float*)d_in[22];  // [1,512]
  const float* v_b    = (const float*)d_in[23];
  const float* wh_w   = (const float*)d_in[24];  // [1,1024]
  const float* wh_b   = (const float*)d_in[25];
  const float* ws_w   = (const float*)d_in[26];  // [1,512]
  const float* ws_b   = (const float*)d_in[27];
  const float* wx_w   = (const float*)d_in[28];  // [1,256]
  const float* wx_b   = (const float*)d_in[29];

  float* out = (float*)d_out;
  float* cov_out = out + (out_size - 1);

  // workspace carve
  char* wp = (char*)d_ws;
  auto alloc = [&](size_t nbytes) -> void* {
    void* p = (void*)wp;
    wp += (nbytes + 255) & ~(size_t)255;
    return p;
  };
  __bf16* enc_xbf   = (__bf16*)alloc((size_t)S_ * B_ * E_ * 2);
  __bf16* dec_xbf   = (__bf16*)alloc((size_t)T_ * B_ * E_ * 2);
  __bf16* efWih_bf  = (__bf16*)alloc((size_t)G_ * E_ * 2);
  __bf16* ebWih_bf  = (__bf16*)alloc((size_t)G_ * E_ * 2);
  __bf16* dWih_bf   = (__bf16*)alloc((size_t)G_ * E_ * 2);
  __bf16* efWhh_bf  = (__bf16*)alloc((size_t)G_ * H_ * 2);
  __bf16* ebWhh_bf  = (__bf16*)alloc((size_t)G_ * H_ * 2);
  __bf16* dWhh_bf   = (__bf16*)alloc((size_t)G_ * H_ * 2);
  float* hf         = (float*)alloc((size_t)S_ * B_ * H_ * 4);
  float* hb         = (float*)alloc((size_t)S_ * B_ * H_ * 4);
  float* ep         = (float*)alloc((size_t)S_ * B_ * H_ * 4);
  float* epv        = (float*)alloc((size_t)S_ * B_ * 4);
  float* attn       = (float*)alloc((size_t)B_ * S_ * 4);
  float* context    = (float*)alloc((size_t)B_ * 2 * H_ * 4);
  float* pgctx      = (float*)alloc((size_t)B_ * 4);
  float* hidden_all = (float*)alloc((size_t)T_ * B_ * CH3_ * 4);
  float* pgen_all   = (float*)alloc((size_t)T_ * B_ * 4);
  float* tmp1       = (float*)alloc((size_t)T_ * B_ * CH3_ * 4);
  float* logits     = (float*)alloc((size_t)T_ * B_ * V_ * 4);
  float* rowstat    = (float*)alloc((size_t)T_ * B_ * 2 * 4);
  float* p_copy     = (float*)alloc((size_t)B_ * VE_ * 4);
  float* row_tok    = (float*)alloc((size_t)B_ * VE_ * 4);
  int*   counts     = (int*)  alloc((size_t)VE_ * 4);

  // zero the scatter targets
  hipMemsetAsync(p_copy,  0, (size_t)B_ * VE_ * 4, stream);
  hipMemsetAsync(row_tok, 0, (size_t)B_ * VE_ * 4, stream);
  hipMemsetAsync(counts,  0, (size_t)VE_ * 4, stream);

  // 0) bf16 conversions of LSTM weights (few MB, fully parallel)
  const int nIh = G_ * E_, nHh = G_ * H_;
  f32_to_bf16<<<(nIh + 255) / 256, 256, 0, stream>>>(efWih, efWih_bf, nIh);
  f32_to_bf16<<<(nIh + 255) / 256, 256, 0, stream>>>(ebWih, ebWih_bf, nIh);
  f32_to_bf16<<<(nIh + 255) / 256, 256, 0, stream>>>(dWih,  dWih_bf,  nIh);
  f32_to_bf16<<<(nHh + 255) / 256, 256, 0, stream>>>(efWhh, efWhh_bf, nHh);
  f32_to_bf16<<<(nHh + 255) / 256, 256, 0, stream>>>(ebWhh, ebWhh_bf, nHh);
  f32_to_bf16<<<(nHh + 255) / 256, 256, 0, stream>>>(dWhh,  dWhh_bf,  nHh);

  // 1) embeddings (bf16 activations)
  gather_enc<<<(S_ * B_ * E_ + 255) / 256, 256, 0, stream>>>(input, emb, enc_xbf);
  gather_dec<<<(T_ * B_ * E_ + 255) / 256, 256, 0, stream>>>(target, emb, dec_xbf);

  // 2) bidirectional encoder LSTM (bf16 WMMA, both directions concurrently)
  lstm_encoder<<<2, 1024, 0, stream>>>(enc_xbf, efWih_bf, efWhh_bf, efb, hf,
                                       ebWih_bf, ebWhh_bf, ebb, hb);

  // 3) enc_proj = enc_out @ Wh^T + Wh_b  (M=6400, N=512, K=512+512; Wh is 2MB, L2-resident)
  gemm_wt<<<dim3(S_ * B_ / 16, 32 / 8), 256, 0, stream>>>(hf, H_, Wh_w,       2 * H_, Wh_b, ep, H_, 32, H_, 0);
  gemm_wt<<<dim3(S_ * B_ / 16, 32 / 8), 256, 0, stream>>>(hb, H_, Wh_w + H_,  2 * H_, NULL, ep, H_, 32, H_, 1);

  // 4) step-invariant attention (add-term cancels in softmax), closed-form cov_loss
  epv_kernel<<<(S_ * B_ + 7) / 8, 256, 0, stream>>>(ep, v_w, v_b, epv);
  attn_kernel<<<1, 512, 0, stream>>>(epv, attn, cov_out);

  // 5) constant context and its p_gen contribution
  context_kernel<<<(B_ * 2 * H_ + 255) / 256, 256, 0, stream>>>(attn, hf, hb, context);
  pgctx_kernel<<<1, 512, 0, stream>>>(context, wh_w, wh_b, pgctx);

  // 6) constant pointer-copy distribution
  counts_kernel<<<(B_ * S_ + 255) / 256, 256, 0, stream>>>(input, counts);
  rowtok_kernel<<<(B_ * S_ + 255) / 256, 256, 0, stream>>>(input, attn, row_tok);
  pcopy_kernel<<<(B_ * S_ + 255) / 256, 256, 0, stream>>>(input, counts, attn, row_tok, p_copy);

  // 7) sequential decoder LSTM (bf16 WMMA) -> hidden_all [512][1536], pgen_all
  decoder_seq<<<1, 1024, 0, stream>>>(dec_xbf, dWih_bf, dWhh_bf, db, ws_w, ws_b, wx_w, wx_b,
                                      pgctx, context, hidden_all, pgen_all);

  // 8) hoisted vocab head, strip GEMMs: each W element streamed from HBM exactly once
  gemm_strip<<<CH3_ / 16, 256, 0, stream>>>(hidden_all, CH3_, V1_w, CH3_, V1_b, tmp1, CH3_, CH3_);
  gemm_strip<<<V_ / 16,   256, 0, stream>>>(tmp1,       CH3_, V2_w, CH3_, V2_b, logits, V_,  CH3_);

  // 9) softmax stats + final mixture into d_out
  rowstat_kernel<<<T_ * B_, 256, 0, stream>>>(logits, rowstat);
  size_t total = (size_t)B_ * T_ * VE_;
  combine_kernel<<<(unsigned)((total + 255) / 256), 256, 0, stream>>>(logits, rowstat, p_copy, pgen_all, out);
}
